// Fusion_Cross_Conv_Gate_62500364091874
// MI455X (gfx1250) — matrix operand order repaired and checked by reading, not compiled, not run on hardware
//
#include <hip/hip_runtime.h>
#include <hip/hip_bf16.h>
#include <math.h>

#define BB 4
#define NN 32768
#define HH 96
#define WW 320
#define IC 128
#define PC 128
#define OCH 256
#define RC 32
#define FC (IC + PC)        // 256
#define HWS (HH * WW)       // 30720
#define K2 (FC * 9)         // 2304
#define EPSV 1e-5f
#define CSTR 264            // padded channel stride for conv2 LDS tile
#define TCOLS 34            // 32-pixel segment + 1 halo col each side

typedef _Float16 f16;
typedef __attribute__((ext_vector_type(16))) _Float16 v16h;
typedef __attribute__((ext_vector_type(8)))  _Float16 v8h;
typedef __attribute__((ext_vector_type(8)))  float    v8f;

static __device__ __forceinline__ v16h cat8(v8h a, v8h b) {
  return __builtin_shufflevector(a, b, 0,1,2,3,4,5,6,7,8,9,10,11,12,13,14,15);
}

// ---------------- zero scatter target + stats ----------------
__global__ void k_zero(float* __restrict__ p2i, float* __restrict__ st1, float* __restrict__ st2) {
  const long total = (long)BB * PC * HWS;
  for (long i = (long)blockIdx.x * blockDim.x + threadIdx.x; i < total;
       i += (long)gridDim.x * blockDim.x)
    p2i[i] = 0.f;
  int t = blockIdx.x * blockDim.x + threadIdx.x;
  if (t < 2 * PC) st1[t] = 0.f;
  if (t < 2 * OCH) st2[t] = 0.f;
}

// ---------------- gating MLP: att[b,n] ----------------
__global__ __launch_bounds__(256) void k_gate(
    const float* __restrict__ pfeat, const float* __restrict__ ifeat,
    const float* __restrict__ fc1w, const float* __restrict__ fc1b,
    const float* __restrict__ fc2w, const float* __restrict__ fc2b,
    const float* __restrict__ fc3w, const float* __restrict__ fc3b,
    float* __restrict__ att) {
  __shared__ float w1[RC * IC];
  __shared__ float w2[RC * PC];
  __shared__ float b12[RC];
  __shared__ float w3[RC];
  for (int i = threadIdx.x; i < RC * IC; i += blockDim.x) w1[i] = fc1w[i];
  for (int i = threadIdx.x; i < RC * PC; i += blockDim.x) w2[i] = fc2w[i];
  if (threadIdx.x < RC) {
    b12[threadIdx.x] = fc1b[threadIdx.x] + fc2b[threadIdx.x];
    w3[threadIdx.x] = fc3w[threadIdx.x];
  }
  __syncthreads();
  int gid = blockIdx.x * blockDim.x + threadIdx.x;   // 0 .. B*N-1
  int b = gid / NN, n = gid % NN;
  float acc[RC];
#pragma unroll
  for (int r = 0; r < RC; r++) acc[r] = b12[r];
  const float* ip = ifeat + (long)b * IC * NN + n;
#pragma unroll 2
  for (int c = 0; c < IC; c++) {
    float x = ip[(long)c * NN];
#pragma unroll
    for (int r = 0; r < RC; r++) acc[r] = fmaf(w1[r * IC + c], x, acc[r]);
  }
  const float* pp = pfeat + (long)b * PC * NN + n;
#pragma unroll 2
  for (int c = 0; c < PC; c++) {
    float x = pp[(long)c * NN];
#pragma unroll
    for (int r = 0; r < RC; r++) acc[r] = fmaf(w2[r * PC + c], x, acc[r]);
  }
  float s = fc3b[0];
#pragma unroll
  for (int r = 0; r < RC; r++) s = fmaf(w3[r], tanhf(acc[r]), s);
  att[gid] = 1.f / (1.f + expf(-s));
}

// ---------------- conv1d (k=1) as WMMA GEMM: pf = W[128x128] @ X[128xN] + b ----------------
__global__ __launch_bounds__(256) void k_pf_gemm(
    const float* __restrict__ pfeat, const float* __restrict__ w,
    const float* __restrict__ bias, float* __restrict__ pf) {
  __shared__ f16 Wl[PC * PC];   // [o][k], k fastest
  __shared__ f16 Xl[64 * PC];   // [p][k], k fastest
  int tid = threadIdx.x;
  int bid = blockIdx.x;
  int b = bid / (NN / 64);
  int n0 = (bid % (NN / 64)) * 64;
  for (int i = tid; i < PC * PC; i += 256) Wl[i] = (f16)w[i];
  for (int i = tid; i < 64 * PC; i += 256) {
    int k = i / 64, p = i % 64;                       // consecutive tid -> consecutive n
    Xl[p * PC + k] = (f16)pfeat[((long)b * PC + k) * NN + n0 + p];
  }
  __syncthreads();
  int lane = tid & 31, wv = tid >> 5;
  int half = lane >> 4, lr = lane & 15;
  int o0 = wv * 16;
  v8f acc0 = {}, acc1 = {}, acc2 = {}, acc3 = {};
#pragma unroll
  for (int k0 = 0; k0 < PC; k0 += 32) {
    const v8h* a0 = (const v8h*)&Wl[(o0 + lr) * PC + k0 + half * 16];
    const v8h* a1 = (const v8h*)&Wl[(o0 + lr) * PC + k0 + half * 16 + 8];
    v16h af = cat8(*a0, *a1);
    const v8h* b0 = (const v8h*)&Xl[(0 + lr) * PC + k0 + half * 16];
    const v8h* b1 = (const v8h*)&Xl[(16 + lr) * PC + k0 + half * 16];
    const v8h* b2 = (const v8h*)&Xl[(32 + lr) * PC + k0 + half * 16];
    const v8h* b3 = (const v8h*)&Xl[(48 + lr) * PC + k0 + half * 16];
    v16h bf0 = cat8(b0[0], b0[1]);
    v16h bf1 = cat8(b1[0], b1[1]);
    v16h bf2 = cat8(b2[0], b2[1]);
    v16h bf3 = cat8(b3[0], b3[1]);
    acc0 = __builtin_amdgcn_wmma_f32_16x16x32_f16(false, af, false, bf0, (short)0, acc0, false, false);
    acc1 = __builtin_amdgcn_wmma_f32_16x16x32_f16(false, af, false, bf1, (short)0, acc1, false, false);
    acc2 = __builtin_amdgcn_wmma_f32_16x16x32_f16(false, af, false, bf2, (short)0, acc2, false, false);
    acc3 = __builtin_amdgcn_wmma_f32_16x16x32_f16(false, af, false, bf3, (short)0, acc3, false, false);
  }
#pragma unroll
  for (int v = 0; v < 8; v++) {
    int o = o0 + v + half * 8;                 // C/D layout: VGPR v, halves hold M=v and M=v+8
    float bv = bias[o];
    long rb = ((long)b * PC + o) * NN + n0 + lr;
    pf[rb + 0]  = acc0[v] + bv;
    pf[rb + 16] = acc1[v] + bv;
    pf[rb + 32] = acc2[v] + bv;
    pf[rb + 48] = acc3[v] + bv;
  }
}

// ---------------- BN1d stats: one block per channel ----------------
__global__ __launch_bounds__(256) void k_stats1(const float* __restrict__ pf, float* __restrict__ st1) {
  __shared__ float s1[256], s2[256];
  int o = blockIdx.x;
  float a = 0.f, q = 0.f;
  for (int b = 0; b < BB; b++) {
    const float* p = pf + ((long)b * PC + o) * NN;
    for (int n = threadIdx.x; n < NN; n += 256) { float v = p[n]; a += v; q += v * v; }
  }
  s1[threadIdx.x] = a; s2[threadIdx.x] = q; __syncthreads();
  for (int st = 128; st > 0; st >>= 1) {
    if ((int)threadIdx.x < st) { s1[threadIdx.x] += s1[threadIdx.x + st]; s2[threadIdx.x] += s2[threadIdx.x + st]; }
    __syncthreads();
  }
  if (threadIdx.x == 0) { st1[o] = s1[0]; st1[PC + o] = s2[0]; }
}

__global__ void k_fin1(const float* __restrict__ st1, const float* __restrict__ g,
                       const float* __restrict__ bt, float* __restrict__ bn1) {
  int o = threadIdx.x;
  float cnt = (float)((long)BB * NN);
  float mu = st1[o] / cnt;
  float var = st1[PC + o] / cnt - mu * mu;
  float sc = g[o] * rsqrtf(var + EPSV);
  bn1[o] = sc;
  bn1[PC + o] = bt[o] - mu * sc;
}

// ---------------- bilinear scatter (reverse grid_sample) ----------------
__global__ __launch_bounds__(128) void k_scatter(
    const float* __restrict__ pf, const float* __restrict__ att,
    const float* __restrict__ xy, const float* __restrict__ bn1,
    float* __restrict__ p2i) {
  int pid = blockIdx.x;                     // b*N + n
  int b = pid / NN, n = pid % NN;
  float ax = xy[(long)pid * 2 + 0], ay = xy[(long)pid * 2 + 1];
  float ix = ((ax + 1.f) * WW - 1.f) * 0.5f;
  float iy = ((ay + 1.f) * HH - 1.f) * 0.5f;
  float x0 = floorf(ix), y0 = floorf(iy);
  float wx1 = ix - x0, wx0 = 1.f - wx1, wy1 = iy - y0, wy0 = 1.f - wy1;
  int xi0 = (int)x0, yi0 = (int)y0, xi1 = xi0 + 1, yi1 = yi0 + 1;
  float a = att[pid];
  int o = threadIdx.x;
  float v = pf[((long)b * PC + o) * NN + n] * bn1[o] + bn1[PC + o];
  v = fmaxf(v, 0.f) * a;
  float* base = p2i + ((long)b * PC + o) * HWS;
  if (xi0 >= 0 && xi0 < WW && yi0 >= 0 && yi0 < HH) atomicAdd(base + yi0 * WW + xi0, v * wx0 * wy0);
  if (xi1 >= 0 && xi1 < WW && yi0 >= 0 && yi0 < HH) atomicAdd(base + yi0 * WW + xi1, v * wx1 * wy0);
  if (xi0 >= 0 && xi0 < WW && yi1 >= 0 && yi1 < HH) atomicAdd(base + yi1 * WW + xi0, v * wx0 * wy1);
  if (xi1 >= 0 && xi1 < WW && yi1 >= 0 && yi1 < HH) atomicAdd(base + yi1 * WW + xi1, v * wx1 * wy1);
}

// ---------------- pack conv2d weights: f32 [oc][c][3][3] -> f16 [oc][tap*256+c] ----------------
__global__ void k_wpack(const float* __restrict__ w, f16* __restrict__ wpk) {
  long total = (long)OCH * K2;
  for (long i = (long)blockIdx.x * blockDim.x + threadIdx.x; i < total;
       i += (long)gridDim.x * blockDim.x) {
    int oc = (int)(i / K2);
    int r = (int)(i % K2);
    int tap = r / FC, c = r % FC;
    wpk[i] = (f16)w[((long)oc * FC + c) * 9 + tap];
  }
}

// ---------------- concat(p2i, image) -> f16 fusion [b][c][hw] ----------------
__global__ void k_fuse(const float* __restrict__ p2i, const float* __restrict__ img,
                       f16* __restrict__ fus) {
  long total = (long)BB * FC * HWS;
  for (long i = (long)blockIdx.x * blockDim.x + threadIdx.x; i < total;
       i += (long)gridDim.x * blockDim.x) {
    int hw = (int)(i % HWS);
    long t = i / HWS;
    int c = (int)(t % FC);
    int b = (int)(t / FC);
    float v = (c < PC) ? p2i[((long)b * PC + c) * HWS + hw]
                       : img[((long)b * IC + (c - PC)) * HWS + hw];
    fus[i] = (f16)v;
  }
}

// ---------------- 3x3 conv as implicit WMMA GEMM, raw (pre-BN) output ----------------
// Block: 512 threads = 16 waves; each wave = one 16-oc tile over a 32-pixel row segment.
// Each A (weight) fragment feeds 2 WMMAs (two 16-pixel subtiles) -> halves weight traffic.
__global__ __launch_bounds__(512) void k_conv2(
    const f16* __restrict__ fus, const f16* __restrict__ wpk, float* __restrict__ y) {
  __shared__ f16 T[3 * TCOLS * CSTR];       // [row][col][c], padded c-stride (~52.6 KB)
  int tid = threadIdx.x;
  int bid = blockIdx.x;
  const int SEGS = WW / 32;                 // 10
  int b = bid / (HH * SEGS);
  int rem = bid % (HH * SEGS);
  int h = rem / SEGS;
  int w0 = (rem % SEGS) * 32;
  // stage 3 rows x 34 cols x 256 channels, zero pad at borders
  for (int i = tid; i < 3 * TCOLS * FC; i += 512) {
    int col = i % TCOLS;
    int t2 = i / TCOLS;
    int c = t2 % FC;
    int row = t2 / FC;
    int hh = h - 1 + row, wp = w0 - 1 + col;
    f16 v = (f16)0.f;
    if (hh >= 0 && hh < HH && wp >= 0 && wp < WW)
      v = fus[((long)b * FC + c) * HWS + hh * WW + wp];
    T[(row * TCOLS + col) * CSTR + c] = v;
  }
  __syncthreads();
  int lane = tid & 31, wv = tid >> 5;       // 16 waves: one 16-oc tile each
  int half = lane >> 4, lr = lane & 15;
  int oc0 = wv * 16;
  v8f accA = {}, accB = {};
  const f16* wrow = wpk + (long)(oc0 + lr) * K2;
  __builtin_prefetch(wrow, 0, 3);           // global_prefetch_b8
#pragma unroll 1
  for (int tap = 0; tap < 9; tap++) {
    int ky = tap / 3, kx = tap % 3;
    const f16* wt = wrow + tap * FC;
    const f16* tbA = &T[(ky * TCOLS + kx + lr) * CSTR];        // pixels w0 .. w0+15
    const f16* tbB = &T[(ky * TCOLS + kx + lr + 16) * CSTR];   // pixels w0+16 .. w0+31
#pragma unroll
    for (int c0 = 0; c0 < FC; c0 += 32) {
      const v8h* a0 = (const v8h*)(wt + c0 + half * 16);
      const v8h* a1 = (const v8h*)(wt + c0 + half * 16 + 8);
      v16h af = cat8(*a0, *a1);
      const v8h* bpA = (const v8h*)(tbA + c0 + half * 16);
      const v8h* bpB = (const v8h*)(tbB + c0 + half * 16);
      v16h bfA = cat8(bpA[0], bpA[1]);
      v16h bfB = cat8(bpB[0], bpB[1]);
      accA = __builtin_amdgcn_wmma_f32_16x16x32_f16(false, af, false, bfA, (short)0, accA, false, false);
      accB = __builtin_amdgcn_wmma_f32_16x16x32_f16(false, af, false, bfB, (short)0, accB, false, false);
    }
  }
  long obase = (long)b * OCH * HWS + (long)h * WW + w0 + lr;
#pragma unroll
  for (int v = 0; v < 8; v++) {
    int oc = oc0 + v + half * 8;
    y[obase + (long)oc * HWS + 0]  = accA[v];
    y[obase + (long)oc * HWS + 16] = accB[v];
  }
}

// ---------------- BN2d stats / finalize / apply ----------------
__global__ __launch_bounds__(256) void k_stats2(const float* __restrict__ y, float* __restrict__ st2) {
  __shared__ float s1[256], s2[256];
  int oc = blockIdx.x;
  float a = 0.f, q = 0.f;
  for (int b = 0; b < BB; b++) {
    const float* p = y + ((long)b * OCH + oc) * HWS;
    for (int i = threadIdx.x; i < HWS; i += 256) { float v = p[i]; a += v; q += v * v; }
  }
  s1[threadIdx.x] = a; s2[threadIdx.x] = q; __syncthreads();
  for (int st = 128; st > 0; st >>= 1) {
    if ((int)threadIdx.x < st) { s1[threadIdx.x] += s1[threadIdx.x + st]; s2[threadIdx.x] += s2[threadIdx.x + st]; }
    __syncthreads();
  }
  if (threadIdx.x == 0) { st2[oc] = s1[0]; st2[OCH + oc] = s2[0]; }
}

__global__ void k_fin2(const float* __restrict__ st2, const float* __restrict__ g,
                       const float* __restrict__ bt, float* __restrict__ bn2) {
  int o = threadIdx.x;
  float cnt = (float)((long)BB * HWS);
  float mu = st2[o] / cnt;
  float var = st2[OCH + o] / cnt - mu * mu;
  float sc = g[o] * rsqrtf(var + EPSV);
  bn2[o] = sc;
  bn2[OCH + o] = bt[o] - mu * sc;
}

__global__ void k_bnrelu(float* __restrict__ y, const float* __restrict__ bn2) {
  long total = (long)BB * OCH * HWS;
  for (long i = (long)blockIdx.x * blockDim.x + threadIdx.x; i < total;
       i += (long)gridDim.x * blockDim.x) {
    int oc = (int)((i / HWS) % OCH);
    float v = y[i] * bn2[oc] + bn2[OCH + oc];
    y[i] = fmaxf(v, 0.f);
  }
}

extern "C" void kernel_launch(void* const* d_in, const int* in_sizes, int n_in,
                              void* d_out, int out_size, void* d_ws, size_t ws_size,
                              hipStream_t stream) {
  (void)in_sizes; (void)n_in; (void)out_size; (void)ws_size;
  const float* point_features = (const float*)d_in[0];
  const float* img_features   = (const float*)d_in[1];
  const float* li_xy_cor      = (const float*)d_in[2];
  const float* image          = (const float*)d_in[3];
  const float* fc1_w = (const float*)d_in[4];
  const float* fc1_b = (const float*)d_in[5];
  const float* fc2_w = (const float*)d_in[6];
  const float* fc2_b = (const float*)d_in[7];
  const float* fc3_w = (const float*)d_in[8];
  const float* fc3_b = (const float*)d_in[9];
  const float* conv1d_w = (const float*)d_in[10];
  const float* conv1d_b = (const float*)d_in[11];
  const float* bn1d_g = (const float*)d_in[12];
  const float* bn1d_b = (const float*)d_in[13];
  const float* conv2d_w = (const float*)d_in[14];
  const float* bn2d_g = (const float*)d_in[15];
  const float* bn2d_b = (const float*)d_in[16];
  float* out = (float*)d_out;

  // workspace carve-up (bytes, 256-aligned)
  char* ws = (char*)d_ws;
  size_t off = 0;
  float* pf  = (float*)(ws + off); off += (size_t)BB * PC * NN * 4;            // 64 MB
  float* att = (float*)(ws + off); off += (size_t)BB * NN * 4;                 // 0.5 MB
  float* p2i = (float*)(ws + off); off += (size_t)BB * PC * HWS * 4;           // 60 MB
  f16*  fus  = (f16*)  (ws + off); off += (size_t)BB * FC * HWS * 2;           // 60 MB
  f16*  wpk  = (f16*)  (ws + off); off += (size_t)OCH * K2 * 2;                // 1.2 MB
  float* st1 = (float*)(ws + off); off += 2 * PC * 4;
  float* st2 = (float*)(ws + off); off += 2 * OCH * 4;
  float* bn1 = (float*)(ws + off); off += 2 * PC * 4;
  float* bn2 = (float*)(ws + off); off += 2 * OCH * 4;

  k_zero<<<4096, 256, 0, stream>>>(p2i, st1, st2);
  k_gate<<<(BB * NN) / 256, 256, 0, stream>>>(point_features, img_features,
      fc1_w, fc1_b, fc2_w, fc2_b, fc3_w, fc3_b, att);
  k_pf_gemm<<<(BB * NN) / 64, 256, 0, stream>>>(point_features, conv1d_w, conv1d_b, pf);
  k_stats1<<<PC, 256, 0, stream>>>(pf, st1);
  k_fin1<<<1, PC, 0, stream>>>(st1, bn1d_g, bn1d_b, bn1);
  k_scatter<<<BB * NN, PC, 0, stream>>>(pf, att, li_xy_cor, bn1, p2i);
  k_wpack<<<1024, 256, 0, stream>>>(conv2d_w, wpk);
  k_fuse<<<4096, 256, 0, stream>>>(p2i, image, fus);
  k_conv2<<<BB * HH * (WW / 32), 512, 0, stream>>>(fus, wpk, out);
  k_stats2<<<OCH, 256, 0, stream>>>(out, st2);
  k_fin2<<<1, OCH, 0, stream>>>(st2, bn2d_g, bn2d_b, bn2);
  k_bnrelu<<<4096, 256, 0, stream>>>(out, bn2);
}
